// ALSHConv_7198365188564
// MI455X (gfx1250) — matrix operand order repaired and checked by reading, not compiled, not run on hardware
//
#include <hip/hip_runtime.h>
#include <math.h>

// ---------------- CDNA5 vector types ----------------
typedef __attribute__((ext_vector_type(16))) _Float16 v16h;
typedef __attribute__((ext_vector_type(8)))  _Float16 v8h;
typedef __attribute__((ext_vector_type(8)))  float    v8f;
typedef __attribute__((ext_vector_type(4)))  float    v4f;
typedef __attribute__((ext_vector_type(4)))  unsigned v4u;
typedef __attribute__((ext_vector_type(8)))  int      v8i;
typedef __attribute__((ext_vector_type(4)))  int      v4i;

#ifndef __has_builtin
#define __has_builtin(x) 0
#endif
#if __has_builtin(__builtin_amdgcn_tensor_load_to_lds)
#define HAS_TDM 1
#else
#define HAS_TDM 0
#endif

// ---------------- problem constants ----------------
static constexpr int       BB   = 16;
static constexpr int       CC   = 64;
static constexpr int       HH   = 128;
static constexpr int       WWD  = 128;
static constexpr int       OO   = 256;
static constexpr int       HP   = 130;   // padded H
static constexpr int       WP   = 132;   // padded W (extra col keeps rows even)
static constexpr int       KTOT = 576;   // 64*3*3
static constexpr long long PIX  = (long long)BB * HH * WWD;   // 262144

// =====================================================================
// TDM: pack a 2D tile descriptor (data_size = 2 bytes) and issue the DMA.
// Tile: tile1 rows of tile0 contiguous f16 elements, row stride stride0
// elements in global memory; written densely to LDS (row-major [dim1][dim0]).
// This toolchain's builtin takes 6 args (g0, g1, g2, g3, g4, cpol).
// =====================================================================
#if HAS_TDM
__device__ __forceinline__ void tdm_load_2d(unsigned lds_off, const void* gptr,
                                            unsigned tile0, unsigned tile1,
                                            unsigned long long stride0_elems) {
  unsigned long long ga = (unsigned long long)gptr;
  v4u g0;
  g0[0] = 1u;                                             // count=1, user mode
  g0[1] = lds_off;                                        // lds_addr (bytes)
  g0[2] = (unsigned)(ga & 0xffffffffull);                 // global_addr[31:0]
  g0[3] = (unsigned)((ga >> 32) & 0x01ffffffu) | (2u << 30); // [56:32] | type=2
  // Large tensor dims so OOB never triggers (tiles are in-bounds by padding).
  const unsigned long long d0 = 1ull << 30, d1 = 1ull << 30;
  v8i g1;
  g1[0] = (int)(1u << 16);                                // mask=0, data_size=1 (2B)
  g1[1] = (int)((d0 & 0xffffull) << 16);                  // tensor_dim0[15:0]
  g1[2] = (int)(((d0 >> 16) & 0xffffull) | ((d1 & 0xffffull) << 16));
  g1[3] = (int)(((d1 >> 16) & 0xffffull) | ((unsigned long long)tile0 << 16));
  g1[4] = (int)(tile1 & 0xffffu);                         // tile_dim1, tile_dim2=0
  g1[5] = (int)(stride0_elems & 0xffffffffull);           // dim0_stride[31:0]
  g1[6] = (int)((stride0_elems >> 32) & 0xffffull);       // dim0_stride[47:32]
  g1[7] = 0;
  v4i z4  = {0, 0, 0, 0};
  v8i z8  = {0, 0, 0, 0, 0, 0, 0, 0};
  __builtin_amdgcn_tensor_load_to_lds(g0, g1, z4, z4, z8, 0);
}
#endif

// =====================================================================
// Implicit-GEMM conv kernel:  D[oc, pix] = sum_k W[oc,k] * Patch[k,pix]
//   WM x WN waves, wave tile = 16 pixels x (NT*16) out-channels.
//   K order = (tap, channel) so every K=32 step is 32 contiguous channels
//   of the padded-NHWC f16 input -> one 2D TDM tile per operand per step.
// =====================================================================
template<int WM, int WN, int NT, int NTOT, int MB>
__global__ __launch_bounds__(WM * WN * 32)
void alsh_gemm_kernel(const _Float16* __restrict__ xh,   // [16][130][132][64] f16
                      const _Float16* __restrict__ wt,   // [NTOT][576] f16
                      const float*    __restrict__ mask, // [NTOT] or nullptr
                      float*          __restrict__ out,
                      long long s_oc, long long s_b)
{
  __shared__ __align__(16) _Float16 Ap[2][MB][32];     // patches  [pix][c]
  __shared__ __align__(16) _Float16 Bw[2][NTOT][32];   // weights  [oc][k]

  constexpr int TPR = WWD / MB;
  const int blk = blockIdx.x;
  const int w0  = (blk % TPR) * MB;
  const int row = blk / TPR;
  const int h   = row % HH;
  const int b   = row / HH;

  const int lane   = (int)(threadIdx.x & 31);
  const int waveid = __builtin_amdgcn_readfirstlane((int)(threadIdx.x >> 5));
  const int wm = waveid % WM;
  const int wn = waveid / WM;
  const int hf = lane >> 4;      // which half of the wave
  const int ln = lane & 15;

  v8f acc[NT] = {};
  const int NSTEPS = 18;         // 9 taps * 2 channel-chunks of 32

#if HAS_TDM
  auto issue = [&](int bufi, int s) {
    const int tap = s >> 1;
    const int c0  = (s & 1) * 32;
    const int dh  = tap / 3, dw = tap % 3;
    const _Float16* gp = xh
        + (((long long)b * HP + (h + dh)) * WP + (w0 + dw)) * 64 + c0;
    tdm_load_2d((unsigned)(unsigned long long)&Ap[bufi][0][0],
                gp, 32u, (unsigned)MB, 64ull);
    const _Float16* gw = wt + (long long)s * 32;       // row stride = 576
    tdm_load_2d((unsigned)(unsigned long long)&Bw[bufi][0][0],
                gw, 32u, (unsigned)NTOT, (unsigned long long)KTOT);
  };
  if (waveid == 0) issue(0, 0);
#endif

  for (int s = 0; s < NSTEPS; ++s) {
    const int cur = s & 1;
#if HAS_TDM
    if (waveid == 0) {
      if (s + 1 < NSTEPS) { issue(cur ^ 1, s + 1);
                            __builtin_amdgcn_s_wait_tensorcnt(2); }
      else                { __builtin_amdgcn_s_wait_tensorcnt(0); }
    }
    __syncthreads();
#else
    __syncthreads();
    {
      const int tap = s >> 1, c0 = (s & 1) * 32;
      const int dh = tap / 3, dw = tap % 3;
      for (int i = (int)threadIdx.x; i < MB * 32; i += WM * WN * 32) {
        int p = i >> 5, c = i & 31;
        Ap[cur][p][c] =
            xh[(((long long)b * HP + (h + dh)) * WP + (w0 + dw + p)) * 64 + c0 + c];
      }
      for (int i = (int)threadIdx.x; i < NTOT * 32; i += WM * WN * 32) {
        int n = i >> 5, kk = i & 31;
        Bw[cur][n][kk] = wt[(long long)n * KTOT + s * 32 + kk];
      }
    }
    __syncthreads();
#endif

    // A fragment (16x32 f16): lane<16 -> row=ln, K = {0..7, 16..23};
    //                         lane>=16 -> row=ln, K = {8..15, 24..31}.
    const _Float16* ap = &Ap[cur][wm * 16 + ln][hf ? 8 : 0];
    v8h a0 = *(const v8h*)ap;
    v8h a1 = *(const v8h*)(ap + 16);
    v16h afrag = __builtin_shufflevector(a0, a1,
        0,1,2,3,4,5,6,7,8,9,10,11,12,13,14,15);
#pragma unroll
    for (int t = 0; t < NT; ++t) {
      // B fragment (32x16 f16): lane<16 -> col=ln, K = {0..7, 16..23}; lane>=16 -> K = {8..15, 24..31}.
      const _Float16* bp = &Bw[cur][wn * (NT * 16) + t * 16 + ln][hf ? 8 : 0];
      v8h b0 = *(const v8h*)bp;
      v8h b1 = *(const v8h*)(bp + 16);
      v16h bfrag = __builtin_shufflevector(b0, b1,
          0,1,2,3,4,5,6,7,8,9,10,11,12,13,14,15);
      acc[t] = __builtin_amdgcn_wmma_f32_16x16x32_f16(
          false, afrag, false, bfrag, (short)0, acc[t], false, false);
    }
    __syncthreads();
  }

  // Epilogue: D VGPR r holds (M = r + 8*hf, N = ln); M maps to consecutive w.
  const long long pcol  = (long long)h * WWD + (w0 + wm * 16 + hf * 8);
  const long long bbase = (long long)b * s_b + pcol;
#pragma unroll
  for (int t = 0; t < NT; ++t) {
    const int oc   = wn * (NT * 16) + t * 16 + ln;
    const float mv = mask ? mask[oc] : 1.0f;
    float* op = out + (long long)oc * s_oc + bbase;
    v4f lo = { acc[t][0]*mv, acc[t][1]*mv, acc[t][2]*mv, acc[t][3]*mv };
    v4f hi = { acc[t][4]*mv, acc[t][5]*mv, acc[t][6]*mv, acc[t][7]*mv };
    *(v4f*)op       = lo;
    *(v4f*)(op + 4) = hi;
  }
}

// =====================================================================
// Stage kernels
// =====================================================================
__global__ void alsh_init(unsigned* mtab, int* counts) {
  int t = (int)threadIdx.x;
  if (t < 128)              mtab[t]       = 0u;   // 16 buckets * 8 words
  if (t >= 128 && t < 256)  counts[t-128] = 0;    // 8 hashes * 16 buckets
}

__global__ void alsh_norms(const float* __restrict__ kern,
                           float* __restrict__ normv, float* __restrict__ denom) {
  __shared__ float red[256];
  int o = (int)threadIdx.x;
  const float* kr = kern + (long long)o * KTOT;
  float ss = 0.f;
  for (int j = 0; j < KTOT; ++j) { float v = kr[j]; ss += v * v; }
  float n = sqrtf(ss);
  normv[o] = n;
  red[o] = n;
  __syncthreads();
  for (int s = 128; s > 0; s >>= 1) {
    if (o < s) red[o] = fmaxf(red[o], red[o + s]);
    __syncthreads();
  }
  if (o == 0) *denom = red[0];
}

// weights f32 [O][C][3][3] -> f16 [O][tap*64 + c]  (K = tap-major)
__global__ void alsh_pack_w(const float* __restrict__ kern, _Float16* __restrict__ wh) {
  int o = (int)blockIdx.x, c = (int)threadIdx.x;
  for (int tap = 0; tap < 9; ++tap)
    wh[(long long)o * KTOT + tap * 64 + c] =
        (_Float16)kern[((long long)o * 64 + c) * 9 + tap];
}

// vote weights: rows 0..7 = (U/denom) * a[k][c][tap] ; rows 8..15 = 0
__global__ void alsh_pack_a(const float* __restrict__ av,
                            const float* __restrict__ denom, _Float16* __restrict__ ah) {
  int k = (int)blockIdx.x, c = (int)threadIdx.x;
  float s = 0.99f / *denom;
  for (int tap = 0; tap < 9; ++tap)
    ah[(long long)k * KTOT + tap * 64 + c] =
        (k < 8) ? (_Float16)(s * av[(long long)k * 594 + c * 9 + tap]) : (_Float16)0.f;
}

// hash each kernel row into 8 buckets, build bucket membership bitmask
__global__ void alsh_hash(const float* __restrict__ kern, const float* __restrict__ av,
                          const float* __restrict__ normv, const float* __restrict__ denom,
                          unsigned* __restrict__ mtab) {
  int o = (int)threadIdx.x;
  float s = 0.99f / *denom;
  const float* kr = kern + (long long)o * KTOT;
  for (int k = 0; k < 8; ++k) {
    const float* ak = av + (long long)k * 594;
    float dot = 0.f;
    for (int j = 0; j < KTOT; ++j) dot += s * kr[j] * ak[j];  // kflat order == a order
    float nr = s * normv[o];
    float p = nr * nr;                                        // nrm^(2^(i+1))
    for (int i = 0; i < 9; ++i) { dot += p * ak[576 + i]; p = p * p; }
    for (int i = 0; i < 9; ++i)   dot += 0.5f * ak[585 + i];
    int idx = ((int)fabsf(floorf(dot))) & 15;
    atomicOr(&mtab[idx * 8 + (o >> 5)], 1u << (o & 31));
  }
}

// x f32 NCHW -> f16 padded NHWC [b][hp][wp][c], zero borders
__global__ void alsh_pack_x(const float* __restrict__ x, _Float16* __restrict__ xh) {
  long long i = (long long)blockIdx.x * 256 + threadIdx.x;
  int c  = (int)(i & 63);
  long long r = i >> 6;
  int wp = (int)(r % WP);
  long long r2 = r / WP;
  int hp = (int)(r2 % HP);
  int b  = (int)(r2 / HP);
  float v = 0.f;
  if (hp >= 1 && hp <= HH && wp >= 1 && wp <= WWD)
    v = x[(((long long)b * CC + c) * HH + (hp - 1)) * WWD + (wp - 1)];
  xh[i] = (_Float16)v;
}

// per-pixel bucket vote + histogram; adds analytic 0.5-plane border term
__global__ void alsh_bucket(const float* __restrict__ dotted,
                            const float* __restrict__ av, int* __restrict__ counts) {
  __shared__ int hist[8 * 16];
  int tid = (int)threadIdx.x;
  if (tid < 128) hist[tid] = 0;
  __syncthreads();
  long long pix = (long long)blockIdx.x * 256 + tid;
  int w = (int)(pix & 127), h = (int)((pix >> 7) & 127);
  for (int k = 0; k < 8; ++k) {
    float d = dotted[(long long)k * PIX + pix];
    float e = 0.f;
    for (int dh = 0; dh < 3; ++dh) {
      int hh = h + dh - 1; if (hh < 0 || hh >= HH) continue;
      for (int dw = 0; dw < 3; ++dw) {
        int ww = w + dw - 1; if (ww < 0 || ww >= WWD) continue;
        e += av[(long long)k * 594 + 576 + dh * 3 + dw];   // channel-64 taps
      }
    }
    d += 0.5f * e;
    int bkt = ((int)fabsf(floorf(d))) & 15;
    atomicAdd(&hist[k * 16 + bkt], 1);
  }
  __syncthreads();
  if (tid < 128) atomicAdd(&counts[tid], hist[tid]);
}

// argmax per hash, union of bucket rows -> float activity mask [256]
__global__ void alsh_finalize(const int* __restrict__ counts,
                              const unsigned* __restrict__ mtab, float* __restrict__ mask) {
  __shared__ int chosen[8];
  __shared__ unsigned aw[8];
  int t = (int)threadIdx.x;
  if (t < 8) {
    int best = counts[t * 16], bi = 0;
    for (int i = 1; i < 16; ++i) {
      int c = counts[t * 16 + i];
      if (c > best) { best = c; bi = i; }      // strict > keeps first max (argmax)
    }
    chosen[t] = bi;
  }
  __syncthreads();
  if (t < 8) {
    unsigned wrd = 0;
    for (int k = 0; k < 8; ++k) wrd |= mtab[chosen[k] * 8 + t];
    aw[t] = wrd;
  }
  __syncthreads();
  for (int o = t; o < 256; o += 64)
    mask[o] = ((aw[o >> 5] >> (o & 31)) & 1u) ? 1.0f : 0.0f;
}

// =====================================================================
extern "C" void kernel_launch(void* const* d_in, const int* in_sizes, int n_in,
                              void* d_out, int out_size, void* d_ws, size_t ws_size,
                              hipStream_t stream) {
  const float* x    = (const float*)d_in[0];
  const float* kern = (const float*)d_in[1];
  const float* av   = (const float*)d_in[2];
  float* out = (float*)d_out;
  char*  ws  = (char*)d_ws;

  size_t curoff = 0;
  auto take = [&](size_t bytes) {
    size_t o = curoff; curoff += (bytes + 255) & ~(size_t)255; return o;
  };
  _Float16* xh     = (_Float16*)(ws + take((size_t)BB * HP * WP * 64 * 2)); // 35.1 MB
  _Float16* wh     = (_Float16*)(ws + take((size_t)OO * KTOT * 2));
  _Float16* ah     = (_Float16*)(ws + take((size_t)16 * KTOT * 2));
  float*    dotted = (float*)   (ws + take((size_t)16 * PIX * 4));          // 16.8 MB
  float*    normv  = (float*)   (ws + take(256 * 4));
  float*    denom  = (float*)   (ws + take(256));
  unsigned* mtab   = (unsigned*)(ws + take(16 * 8 * 4));
  int*      counts = (int*)     (ws + take(8 * 16 * 4));
  float*    mask   = (float*)   (ws + take(256 * 4));

  alsh_init    <<<1,   256, 0, stream>>>(mtab, counts);
  alsh_norms   <<<1,   256, 0, stream>>>(kern, normv, denom);
  alsh_pack_w  <<<OO,  64,  0, stream>>>(kern, wh);
  alsh_pack_a  <<<16,  64,  0, stream>>>(av, denom, ah);
  alsh_hash    <<<1,   256, 0, stream>>>(kern, av, normv, denom, mtab);
  alsh_pack_x  <<<(BB * HP * WP * 64) / 256, 256, 0, stream>>>(x, xh);

  // vote conv: N=16 (8 real hashes + 8 zero rows), out layout [k][b][h][w]
  alsh_gemm_kernel<8, 1, 1, 16, 128>
      <<<BB * HH * (WWD / 128), 256, 0, stream>>>(xh, ah, nullptr, dotted,
                                                  PIX, (long long)HH * WWD);
  alsh_bucket  <<<(int)(PIX / 256), 256, 0, stream>>>(dotted, av, counts);
  alsh_finalize<<<1,   64,  0, stream>>>(counts, mtab, mask);

  // main conv: N=256, out layout [b][o][h][w], masked in epilogue
  alsh_gemm_kernel<2, 4, 4, 256, 32>
      <<<BB * HH * (WWD / 32), 256, 0, stream>>>(xh, wh, mask, out,
                                                 (long long)HH * WWD,
                                                 (long long)OO * HH * WWD);
  (void)in_sizes; (void)n_in; (void)out_size; (void)ws_size;
}
